// Voxelization_37855841747690
// MI455X (gfx1250) — matrix operand order repaired and checked by reading, not compiled, not run on hardware
//
#include <hip/hip_runtime.h>

// ---------------------------------------------------------------------------
// PVCNN voxelize + 5x conv3d(32->32,3x3x3) + BN(train) + ReLU (+residuals)
// for MI455X (gfx1250). Convs are implicit-GEMM with V_WMMA_F32_16X16X32_BF16.
// One wave computes FOUR 16-voxel x 32-outchan tiles (y..y+3) so every B
// (weight) load is shared by 4 tiles and every A (activation) plane load is
// shared by up to 3 taps: per tile ~54 loads for 54 WMMAs (was 162 loads).
// Weights are staged in LDS (48B row stride -> conflict-floor b128 reads),
// via GLOBAL_LOAD_ASYNC_TO_LDS_B128 when the toolchain exposes it.
// ---------------------------------------------------------------------------

#define R_ 64
#define C_ 32
#define V_ (R_ * R_ * R_)      // 262144 voxels
#define XSZ_ (C_ * V_)         // 8388608 grid elements
#define BN_EPS_ 1e-5f

typedef unsigned short u16;
typedef unsigned int   u32;
typedef __attribute__((ext_vector_type(16))) __bf16 v16bf;
typedef __attribute__((ext_vector_type(8)))  float  v8f;
typedef __attribute__((ext_vector_type(4)))  u32    v4u;

union BV32 { v4u u[2]; v16bf v; };   // 32 bytes = one WMMA bf16 operand / lane

#if defined(__has_builtin)
#if __has_builtin(__builtin_amdgcn_global_load_async_to_lds_b128) && \
    __has_builtin(__builtin_amdgcn_s_wait_asynccnt)
#define ASYNC_CP 1
#endif
#endif

#ifdef ASYNC_CP
// exact parameter types per hipcc diagnostic: int4* in AS1 (global) / AS3 (LDS)
typedef __attribute__((__vector_size__(4 * sizeof(int)))) int vsi4;
typedef __attribute__((address_space(1))) vsi4 as1_vsi4;
typedef __attribute__((address_space(3))) vsi4 as3_vsi4;
#endif

__device__ __forceinline__ u16 f2bf(float f) {  // round-to-nearest-even f32->bf16
    u32 u = __float_as_uint(f);
    u += 0x7FFFu + ((u >> 16) & 1u);
    return (u16)(u >> 16);
}

// ---------------------------------------------------------------------------
__global__ void k_zero(float* __restrict__ p, int n) {
    for (int i = blockIdx.x * blockDim.x + threadIdx.x; i < n;
         i += gridDim.x * blockDim.x) p[i] = 0.f;
}

// sum of coords per axis -> stats[0..2]
__global__ void k_coord_sum(const float* __restrict__ coords, int n,
                            float* __restrict__ stats) {
    __shared__ float sx[256], sy[256], sz[256];
    float ax = 0.f, ay = 0.f, az = 0.f;
    for (int i = blockIdx.x * blockDim.x + threadIdx.x; i < n;
         i += gridDim.x * blockDim.x) {
        ax += coords[3 * i + 0];
        ay += coords[3 * i + 1];
        az += coords[3 * i + 2];
    }
    int t = threadIdx.x;
    sx[t] = ax; sy[t] = ay; sz[t] = az;
    __syncthreads();
    for (int g = 128; g > 0; g >>= 1) {
        if (t < g) { sx[t] += sx[t + g]; sy[t] += sy[t + g]; sz[t] += sz[t + g]; }
        __syncthreads();
    }
    if (t == 0) {
        atomicAdd(&stats[0], sx[0]);
        atomicAdd(&stats[1], sy[0]);
        atomicAdd(&stats[2], sz[0]);
    }
}

// max squared norm of centered coords -> stats[3] (float bits via atomicMax)
__global__ void k_maxnorm(const float* __restrict__ coords, int n,
                          float* __restrict__ stats, float invN) {
    __shared__ float sm[256];
    float mx = stats[0] * invN, my = stats[1] * invN, mz = stats[2] * invN;
    float best = 0.f;
    for (int i = blockIdx.x * blockDim.x + threadIdx.x; i < n;
         i += gridDim.x * blockDim.x) {
        float dx = coords[3 * i + 0] - mx;
        float dy = coords[3 * i + 1] - my;
        float dz = coords[3 * i + 2] - mz;
        float n2 = dx * dx + dy * dy + dz * dz;
        best = fmaxf(best, n2);
    }
    int t = threadIdx.x;
    sm[t] = best;
    __syncthreads();
    for (int g = 128; g > 0; g >>= 1) {
        if (t < g) sm[t] = fmaxf(sm[t], sm[t + g]);
        __syncthreads();
    }
    if (t == 0) atomicMax((u32*)&stats[3], __float_as_uint(sm[0]));
}

// nc = clip(((c-mean)/scale + 0.5)*R, 0, R-1); vox = rint(nc); flat index
__global__ void k_normalize(const float* __restrict__ coords, int n,
                            const float* __restrict__ stats, float invN,
                            float* __restrict__ nc_out, int* __restrict__ flat) {
    float mx = stats[0] * invN, my = stats[1] * invN, mz = stats[2] * invN;
    float scale = 2.f * sqrtf(stats[3]);
    float inv_s = 1.f / scale;
    for (int i = blockIdx.x * blockDim.x + threadIdx.x; i < n;
         i += gridDim.x * blockDim.x) {
        float t0 = (coords[3 * i + 0] - mx) * inv_s + 0.5f;
        float t1 = (coords[3 * i + 1] - my) * inv_s + 0.5f;
        float t2 = (coords[3 * i + 2] - mz) * inv_s + 0.5f;
        t0 = fminf(fmaxf(t0 * (float)R_, 0.f), (float)(R_ - 1));
        t1 = fminf(fmaxf(t1 * (float)R_, 0.f), (float)(R_ - 1));
        t2 = fminf(fmaxf(t2 * (float)R_, 0.f), (float)(R_ - 1));
        nc_out[0 * n + i] = t0;
        nc_out[1 * n + i] = t1;
        nc_out[2 * n + i] = t2;
        int v0 = (int)rintf(t0), v1 = (int)rintf(t1), v2 = (int)rintf(t2);
        flat[i] = (v0 * R_ + v1) * R_ + v2;
    }
}

// scatter-add features into channels-last sums grid + counts
__global__ void k_scatter(const float* __restrict__ features,
                          const int* __restrict__ flat, int n,
                          float* __restrict__ sums, float* __restrict__ counts) {
    long long total = (long long)n * C_;
    for (long long idx = (long long)blockIdx.x * blockDim.x + threadIdx.x;
         idx < total; idx += (long long)gridDim.x * blockDim.x) {
        int p = (int)(idx >> 5);
        int c = (int)(idx & 31);
        int fl = flat[p];
        atomicAdd(&sums[(long long)fl * C_ + c], features[(long long)p * C_ + c]);
        if (c == 0) atomicAdd(&counts[fl], 1.f);
    }
}

// per-voxel average; also emit bf16 copy for the first conv's A operand
__global__ void k_divide(float* __restrict__ sums, const float* __restrict__ counts,
                         u16* __restrict__ gbf) {
    for (int i = blockIdx.x * blockDim.x + threadIdx.x; i < XSZ_;
         i += gridDim.x * blockDim.x) {
        float val = sums[i] / fmaxf(counts[i >> 5], 1.f);
        sums[i] = val;
        gbf[i] = f2bf(val);
    }
}

// pack w[O=32][I=32][27] (fp32) into WMMA B layout (bf16):
// wp[t][nblk][half][n(0..15)][k(0..15)]  with K = half*16+k, N = nblk*16+n
__global__ void k_pack_w(const float* __restrict__ w, u16* __restrict__ wp) {
    int tid = blockIdx.x * blockDim.x + threadIdx.x;
    if (tid >= 27 * 1024) return;
    int k    = tid & 15;
    int nn   = (tid >> 4) & 15;
    int half = (tid >> 8) & 1;
    int nblk = (tid >> 9) & 1;
    int t    = tid >> 10;
    int o = nblk * 16 + nn;
    int i = half * 16 + k;
    wp[tid] = f2bf(w[(o * 32 + i) * 27 + t]);
}

// ---------------------------------------------------------------------------
// Implicit-GEMM conv3d 3x3x3 via V_WMMA_F32_16X16X32_BF16.
// One wave computes 4 tiles: (z, y0+i, xb..xb+15) x 32 out-chans, i=0..3.
// LDS weight layout: row = (t*4 + nblk*2 + half)*16 + n, 48B row stride
// (24 u16), 2x16B chunks per row (k 0..7 / 8..15).
#define WROWS_ (27 * 4 * 16)          // 1728 rows
#define LWP_U16_ (WROWS_ * 24)        // 41472 u16 = 82944 B LDS

__global__ void __launch_bounds__(256)
k_conv_wmma(const u16* __restrict__ gbf, const u16* __restrict__ wp,
            const float* __restrict__ bias, float* __restrict__ out) {
    __shared__ u16 lwp[LWP_U16_];

    // ---- stage packed weights into LDS (3456 x 16B chunks) ----
    for (int ci = threadIdx.x; ci < 3456; ci += 256) {
        int row = ci >> 1, kh = ci & 1;
#ifdef ASYNC_CP
        __builtin_amdgcn_global_load_async_to_lds_b128(
            (as1_vsi4*)(wp + ci * 8),
            (as3_vsi4*)&lwp[row * 24 + kh * 8],
            0, 0);
#else
        v4u d = *(const v4u*)(wp + ci * 8);
        *(v4u*)&lwp[row * 24 + kh * 8] = d;
#endif
    }
#ifdef ASYNC_CP
    __builtin_amdgcn_s_wait_asynccnt(0);
#endif
    __syncthreads();

    int wid  = blockIdx.x * 8 + (threadIdx.x >> 5);   // 4096 wave-groups
    int lane = threadIdx.x & 31;
    int m    = lane & 15;        // A-matrix row (voxel within tile)
    int half = lane >> 4;        // K-subset selector (A and B)
    int n15  = lane & 15;        // N within 16-wide block

    int xb = (wid & 3) << 4;           // x base (0,16,32,48)
    int y0 = ((wid >> 2) & 15) << 2;   // y base (multiple of 4)
    int z  = wid >> 6;

    // Bias folded into accumulators (C/D layout: all VGPRs of a lane share N)
    float b0v = bias[n15], b1v = bias[16 + n15];
    v8f acc[4][2];
#pragma unroll
    for (int i = 0; i < 4; ++i)
#pragma unroll
        for (int r = 0; r < 8; ++r) { acc[i][0][r] = b0v; acc[i][1][r] = b1v; }

    for (int dz = -1; dz <= 1; ++dz) {
        int z2 = z + dz;
        bool zok = (unsigned)z2 < 64u;
        for (int dx = -1; dx <= 1; ++dx) {
            int x2 = xb + m + dx;
            bool xok = (unsigned)x2 < 64u;

            // load 6 y-planes once; plane j = i + dy serves tile i at tap dy
            BV32 a[6];
#pragma unroll
            for (int j = 0; j < 6; ++j) {
                int yp = y0 - 1 + j;
                bool ok = zok & xok & ((unsigned)yp < 64u);
                if (ok) {
                    const v4u* p =
                        (const v4u*)(gbf + (long long)((z2 * 64 + yp) * 64 + x2) * 32);
                    a[j].u[0] = p[2 * half];       // lane<16: K0..7 ; lane>=16: K16..23
                    a[j].u[1] = p[2 * half + 1];   // lane<16: K8..15 ; lane>=16: K24..31
                } else {
                    a[j].u[0] = (v4u)0;
                    a[j].u[1] = (v4u)0;
                }
            }

#pragma unroll
            for (int dy = 0; dy < 3; ++dy) {
                int t = (dz + 1) * 9 + dy * 3 + (dx + 1);
                int r0 = ((t * 4 + half) * 16 + n15) * 24;       // nblk 0
                int r1 = ((t * 4 + 2 + half) * 16 + n15) * 24;   // nblk 1
                BV32 b0, b1;
                b0.u[0] = *(const v4u*)&lwp[r0];
                b0.u[1] = *(const v4u*)&lwp[r0 + 8];
                b1.u[0] = *(const v4u*)&lwp[r1];
                b1.u[1] = *(const v4u*)&lwp[r1 + 8];
#pragma unroll
                for (int i = 0; i < 4; ++i) {
                    acc[i][0] = __builtin_amdgcn_wmma_f32_16x16x32_bf16(
                        false, a[i + dy].v, false, b0.v, (short)0, acc[i][0],
                        false, false);
                    acc[i][1] = __builtin_amdgcn_wmma_f32_16x16x32_bf16(
                        false, a[i + dy].v, false, b1.v, (short)0, acc[i][1],
                        false, false);
                }
            }
        }
    }

    // D layout: lane<16 -> M=r, N=lane ; lane>=16 -> M=r+8, N=lane-16
#pragma unroll
    for (int i = 0; i < 4; ++i) {
        long long obase = ((long long)(z * 64 + (y0 + i)) * 64 + xb) * 32;
#pragma unroll
        for (int r = 0; r < 8; ++r) {
            int Mrow = r + half * 8;
            out[obase + Mrow * 32 + n15]      = acc[i][0][r];
            out[obase + Mrow * 32 + 16 + n15] = acc[i][1][r];
        }
    }
}

// per-channel sum and sum-of-squares over the whole grid -> cs[0..31], cs[32..63]
__global__ void k_chan_stats(const float* __restrict__ Y, float* __restrict__ cs) {
    __shared__ float s1[256], s2[256];
    int t = threadIdx.x;
    int c = t & 31;                  // stride is multiple of 32 -> c fixed per thread
    float a = 0.f, b = 0.f;
    for (int i = blockIdx.x * blockDim.x + t; i < XSZ_;
         i += gridDim.x * blockDim.x) {
        float v = Y[i];
        a += v;
        b += v * v;
    }
    s1[t] = a; s2[t] = b;
    __syncthreads();
    for (int g = 128; g >= 32; g >>= 1) {
        if (t < g) { s1[t] += s1[t + g]; s2[t] += s2[t + g]; }
        __syncthreads();
    }
    if (t < 32) {
        atomicAdd(&cs[c], s1[t]);
        atomicAdd(&cs[32 + c], s2[t]);
    }
}

// y = relu(bn(Y) [+ res]); writes fp32 grid + bf16 recast for the next conv
__global__ void k_bn_apply(const float* __restrict__ Y, const float* __restrict__ cs,
                           const float* __restrict__ g, const float* __restrict__ bt,
                           const float* __restrict__ res,
                           float* __restrict__ outF, u16* __restrict__ outBf) {
    __shared__ float sc[32], sh[32];
    if (threadIdx.x < 32) {
        int c = threadIdx.x;
        float invV = 1.f / (float)V_;
        float mean = cs[c] * invV;
        float var  = cs[32 + c] * invV - mean * mean;
        float s = g[c] * rsqrtf(var + BN_EPS_);
        sc[c] = s;
        sh[c] = bt[c] - mean * s;
    }
    __syncthreads();
    for (int i = blockIdx.x * blockDim.x + threadIdx.x; i < XSZ_;
         i += gridDim.x * blockDim.x) {
        int c = i & 31;
        float val = Y[i] * sc[c] + sh[c];
        if (res) val += res[i];
        val = fmaxf(val, 0.f);
        outF[i]  = val;
        outBf[i] = f2bf(val);
    }
}

// final: relu(bn(Y) + res) transposed to [C][D][H][W] in d_out
__global__ void k_final(const float* __restrict__ Y, const float* __restrict__ cs,
                        const float* __restrict__ g, const float* __restrict__ bt,
                        const float* __restrict__ res, float* __restrict__ out) {
    __shared__ float sc[32], sh[32];
    if (threadIdx.x < 32) {
        int c = threadIdx.x;
        float invV = 1.f / (float)V_;
        float mean = cs[c] * invV;
        float var  = cs[32 + c] * invV - mean * mean;
        float s = g[c] * rsqrtf(var + BN_EPS_);
        sc[c] = s;
        sh[c] = bt[c] - mean * s;
    }
    __syncthreads();
    for (int v = blockIdx.x * blockDim.x + threadIdx.x; v < V_;
         v += gridDim.x * blockDim.x) {
        long long base = (long long)v * 32;
#pragma unroll
        for (int c = 0; c < 32; ++c) {
            float val = Y[base + c] * sc[c] + sh[c] + res[base + c];
            out[(long long)c * V_ + v] = fmaxf(val, 0.f);
        }
    }
}

// ---------------------------------------------------------------------------
extern "C" void kernel_launch(void* const* d_in, const int* in_sizes, int n_in,
                              void* d_out, int out_size, void* d_ws, size_t ws_size,
                              hipStream_t stream) {
    (void)n_in; (void)out_size; (void)ws_size;
    const float* features = (const float*)d_in[0];
    const float* coords   = (const float*)d_in[1];
    const float *w[5], *bb[5], *gg[5], *bt[5];
    for (int k = 0; k < 5; ++k) {
        w[k]  = (const float*)d_in[2 + 4 * k];
        bb[k] = (const float*)d_in[3 + 4 * k];
        gg[k] = (const float*)d_in[4 + 4 * k];
        bt[k] = (const float*)d_in[5 + 4 * k];
    }
    const int N = in_sizes[1] / 3;
    float* out_x  = (float*)d_out;
    float* out_nc = out_x + XSZ_;

    // workspace carve-up (~123 MB)
    char* ws = (char*)d_ws;
    size_t off = 0;
    float* stats = (float*)(ws + off); off += 512;
    int*   flat  = (int*)(ws + off);   off += (size_t)N * 4;
    off = (off + 255) & ~(size_t)255;
    u16*   wp    = (u16*)(ws + off);   off += (size_t)5 * 27648 * 2;
    off = (off + 255) & ~(size_t)255;
    float* counts = (float*)(ws + off); off += (size_t)V_ * 4;
    float* GA = (float*)(ws + off);     off += (size_t)XSZ_ * 4;
    float* GB = (float*)(ws + off);     off += (size_t)XSZ_ * 4;
    float* GC = (float*)(ws + off);     off += (size_t)XSZ_ * 4;
    u16*   Gbf = (u16*)(ws + off);      off += (size_t)XSZ_ * 2;
    float* cs = stats + 8;   // per-channel sum / sumsq (64 floats)

    // ---- init ----
    k_zero<<<1, 256, 0, stream>>>(stats, 128);
    k_zero<<<2048, 256, 0, stream>>>(GA, XSZ_);
    k_zero<<<256, 256, 0, stream>>>(counts, V_);
    for (int k = 0; k < 5; ++k)
        k_pack_w<<<108, 256, 0, stream>>>(w[k], wp + (size_t)k * 27648);

    // ---- point normalization ----
    k_coord_sum<<<1024, 256, 0, stream>>>(coords, N, stats);
    k_maxnorm<<<1024, 256, 0, stream>>>(coords, N, stats, 1.f / (float)N);
    k_normalize<<<1024, 256, 0, stream>>>(coords, N, stats, 1.f / (float)N,
                                          out_nc, flat);

    // ---- average voxelization (GA = x0) ----
    k_scatter<<<8192, 256, 0, stream>>>(features, flat, N, GA, counts);
    k_divide<<<4096, 256, 0, stream>>>(GA, counts, Gbf);

    // ---- conv0 + BN + ReLU  (X = GB) ----
    k_conv_wmma<<<512, 256, 0, stream>>>(Gbf, wp + 0 * 27648, bb[0], GB);
    k_zero<<<1, 64, 0, stream>>>(cs, 64);
    k_chan_stats<<<512, 256, 0, stream>>>(GB, cs);
    k_bn_apply<<<4096, 256, 0, stream>>>(GB, cs, gg[0], bt[0], nullptr, GB, Gbf);

    // ---- res block 1 ----
    k_conv_wmma<<<512, 256, 0, stream>>>(Gbf, wp + 1 * 27648, bb[1], GC);
    k_zero<<<1, 64, 0, stream>>>(cs, 64);
    k_chan_stats<<<512, 256, 0, stream>>>(GC, cs);
    k_bn_apply<<<4096, 256, 0, stream>>>(GC, cs, gg[1], bt[1], nullptr, GC, Gbf);

    k_conv_wmma<<<512, 256, 0, stream>>>(Gbf, wp + 2 * 27648, bb[2], GA);
    k_zero<<<1, 64, 0, stream>>>(cs, 64);
    k_chan_stats<<<512, 256, 0, stream>>>(GA, cs);
    k_bn_apply<<<4096, 256, 0, stream>>>(GA, cs, gg[2], bt[2], GB, GA, Gbf); // X = GA

    // ---- res block 2 ----
    k_conv_wmma<<<512, 256, 0, stream>>>(Gbf, wp + 3 * 27648, bb[3], GC);
    k_zero<<<1, 64, 0, stream>>>(cs, 64);
    k_chan_stats<<<512, 256, 0, stream>>>(GC, cs);
    k_bn_apply<<<4096, 256, 0, stream>>>(GC, cs, gg[3], bt[3], nullptr, GC, Gbf);

    k_conv_wmma<<<512, 256, 0, stream>>>(Gbf, wp + 4 * 27648, bb[4], GB);
    k_zero<<<1, 64, 0, stream>>>(cs, 64);
    k_chan_stats<<<512, 256, 0, stream>>>(GB, cs);
    k_final<<<1024, 256, 0, stream>>>(GB, cs, gg[4], bt[4], GA, out_x);
}
